// ADAM_31645319037124
// MI455X (gfx1250) — matrix operand order repaired and checked by reading, not compiled, not run on hardware
//
#include <hip/hip_runtime.h>
#include <hip/hip_bf16.h>
#include <math.h>

// ---------------------------------------------------------------------------
// Fused block-masked attention for MI455X (gfx1250, wave32, WMMA f16->f32).
// B=32, L=641 (padded to 672 = 42*16), C=768, H=12, hd=64, MEM=128.
// ---------------------------------------------------------------------------

typedef __attribute__((ext_vector_type(16))) _Float16 v16h;
typedef __attribute__((ext_vector_type(8)))  _Float16 v8h;
typedef __attribute__((ext_vector_type(8)))  float    v8f;

#define BATCH 32
#define LSEQ  641
#define LP    672            // padded L, multiple of 32
#define CDIM  768
#define NH    12
#define HD    64
#define MEMK  128
#define QKV   (3 * CDIM)
#define NT16  (LP / 16)      // 42
#define NT32  (LP / 32)      // 21
#define SCALE 0.125f         // 1/sqrt(64)

// Workspace layout (bytes, all 256B aligned)
#define OFF_XH   0u
#define OFF_WAT  33030144u   // 32*672*768*2
#define OFF_WPT  36569088u   // + 2304*768*2
#define OFF_QH   37748736u   // + 768*768*2
#define OFF_KH   70778880u   // + 32*12*672*64*2
#define OFF_VT   103809024u
#define OFF_Y    136839168u

#if defined(__HIP_DEVICE_COMPILE__) && defined(__gfx1250__)
#define CDNA5_DEVICE 1
#endif

// 16-bit A-matrix 16x32 per-lane layout (ISA 7.12.2):
// lanes 0-15: row m=lane, halves [0..7]=K0..7, [8..15]=K16..23
// lanes 16-31: row m=lane-16, halves [0..7]=K8..15, [8..15]=K24..31
__device__ inline v16h load_a16(const _Float16* row, int koffA) {
  v16h a;
  *(v8h*)&a       = *(const v8h*)(row + koffA);
  *((v8h*)&a + 1) = *(const v8h*)(row + 16 + koffA);
  return a;
}

// 16-bit B-matrix 32x16: lane n holds column n; lanes 0-15 hold K=0..7,16..23,
// lanes 16-31 hold K=8..15,24..31 (same K order as A).
// `col` points at the column's K-major data (column-major storage).
__device__ inline v16h load_b16(const _Float16* col, int koffB) {
  v16h b;
  *(v8h*)&b       = *(const v8h*)(col + koffB / 2);
  *((v8h*)&b + 1) = *(const v8h*)(col + 16 + koffB / 2);
  return b;
}

// ---------------------------------------------------------------------------
// Conversions
// ---------------------------------------------------------------------------
__global__ __launch_bounds__(256) void k_cvt_x(const float* __restrict__ x,
                                               _Float16* __restrict__ xh) {
  size_t i = (size_t)blockIdx.x * 256 + threadIdx.x;
  size_t total = (size_t)BATCH * LP * CDIM;
  if (i >= total) return;
  int c = (int)(i % CDIM);
  size_t bl = i / CDIM;
  int r = (int)(bl % LP);
  int b = (int)(bl / LP);
  float v = (r < LSEQ) ? x[((size_t)b * LSEQ + r) * CDIM + c] : 0.0f;
  xh[i] = (_Float16)v;
}

__global__ __launch_bounds__(256) void k_cvt_wattn(const float* __restrict__ w,
                                                   _Float16* __restrict__ wt) {
  size_t i = (size_t)blockIdx.x * 256 + threadIdx.x;
  if (i >= (size_t)QKV * CDIM) return;
  int n = (int)(i / CDIM);
  int k = (int)(i % CDIM);
  wt[i] = (_Float16)w[(size_t)k * QKV + n];   // transpose: (N, K) col-major B
}

__global__ __launch_bounds__(256) void k_cvt_wproj(const float* __restrict__ w,
                                                   _Float16* __restrict__ wt) {
  size_t i = (size_t)blockIdx.x * 256 + threadIdx.x;
  if (i >= (size_t)CDIM * CDIM) return;
  int n = (int)(i / CDIM);
  int k = (int)(i % CDIM);
  wt[i] = (_Float16)w[(size_t)k * CDIM + n];
}

// ---------------------------------------------------------------------------
// GEMM1: qkv = x @ W_attn.  One wave per 16x64 tile; scatter to q/k/v(T).
// ---------------------------------------------------------------------------
__global__ __launch_bounds__(32) void k_gemm_qkv(const _Float16* __restrict__ xh,
                                                 const _Float16* __restrict__ wat,
                                                 _Float16* __restrict__ qh,
                                                 _Float16* __restrict__ kh,
                                                 _Float16* __restrict__ vt) {
#ifdef CDNA5_DEVICE
  const int mt = blockIdx.x;          // 0..1343 (B*LP/16)
  const int ng = blockIdx.y;          // 0..35   (2304/64)
  const int lane = threadIdx.x;
  const int m = lane & 15;
  const int koffA = (lane < 16) ? 0 : 8;
  const int koffB = (lane < 16) ? 0 : 16;
  const int mOff  = (lane < 16) ? 0 : 8;

  const _Float16* arow = xh + (size_t)(mt * 16 + m) * CDIM;
  const _Float16* wcol = wat + (size_t)(ng * 64 + m) * CDIM;

  v8f c[4] = {};
#pragma unroll 4
  for (int kk = 0; kk < CDIM; kk += 32) {
    v16h a = load_a16(arow + kk, koffA);
#pragma unroll
    for (int sub = 0; sub < 4; ++sub) {
      v16h b = load_b16(wcol + (size_t)sub * 16 * CDIM + kk, koffB);
      c[sub] = __builtin_amdgcn_wmma_f32_16x16x32_f16(
          false, a, false, b, (short)0, c[sub], false, false);
    }
  }

  const int sec = ng / 12;            // 0=q 1=k 2=v
  const int h   = ng % 12;
#pragma unroll
  for (int sub = 0; sub < 4; ++sub) {
#pragma unroll
    for (int r = 0; r < 8; ++r) {
      const int grow = mt * 16 + r + mOff;
      const int b  = grow / LP;
      const int rr = grow % LP;
      const int d  = sub * 16 + m;
      const _Float16 val = (_Float16)c[sub][r];
      if (sec == 0)
        qh[(((size_t)b * NH + h) * LP + rr) * HD + d] = val;
      else if (sec == 1)
        kh[(((size_t)b * NH + h) * LP + rr) * HD + d] = val;
      else
        vt[(((size_t)b * NH + h) * HD + d) * LP + rr] = val;   // V transposed
    }
  }
#endif
}

// ---------------------------------------------------------------------------
// Fused masked attention: one wave per (b, h, 16-row q tile).
// Scores -> LDS (f32), masked softmax, P (f16) -> LDS, P@V via WMMA.
// ---------------------------------------------------------------------------
__global__ __launch_bounds__(32) void k_attn(const _Float16* __restrict__ qh,
                                             const _Float16* __restrict__ kh,
                                             const _Float16* __restrict__ vt,
                                             _Float16* __restrict__ y) {
#ifdef CDNA5_DEVICE
  __shared__ float    S [16 * LP];   // 43008 B
  __shared__ _Float16 Ph[16 * LP];   // 21504 B

  const int bx = blockIdx.x;
  const int qt = bx % NT16;
  const int bh = bx / NT16;          // b*NH + h
  const int b  = bh / NH;
  const int h  = bh % NH;
  const int lane  = threadIdx.x;
  const int m     = lane & 15;
  const int koffA = (lane < 16) ? 0 : 8;
  const int koffB = (lane < 16) ? 0 : 16;
  const int mOff  = (lane < 16) ? 0 : 8;

  // Q tile (A), hd=64 split into two K=32 chunks, kept in regs for all k-tiles
  const _Float16* qrow = qh + ((size_t)bh * LP + qt * 16 + m) * HD;
  const v16h a0 = load_a16(qrow, koffA);
  const v16h a1 = load_a16(qrow + 32, koffA);

  // ---- pass 1: masked scaled scores into LDS -------------------------------
  for (int t = 0; t < NT16; ++t) {
    const _Float16* krow = kh + ((size_t)bh * LP + t * 16 + m) * HD;
    v16h kb0 = load_b16(krow, koffB);        // d 0..31
    v16h kb1 = load_b16(krow + 32, koffB);   // d 32..63
    v8f c = {};
    c = __builtin_amdgcn_wmma_f32_16x16x32_f16(false, a0, false, kb0, (short)0, c, false, false);
    c = __builtin_amdgcn_wmma_f32_16x16x32_f16(false, a1, false, kb1, (short)0, c, false, false);
#pragma unroll
    for (int r = 0; r < 8; ++r) {
      const int ml = r + mOff;               // local row 0..15
      const int i  = qt * 16 + ml;           // global query row
      const int j  = t * 16 + m;             // global key col
      const bool ok = (j < LSEQ) &&
                      ((i < MEMK) ? (j <= MEMK) : ((i == MEMK) ? true : (j <= i)));
      S[ml * LP + j] = ok ? c[r] * SCALE : -INFINITY;
    }
  }
  __syncthreads();

  // ---- pass 2: softmax (2 lanes per row, halves joined via shfl_xor 16) ----
  {
    const int row  = lane & 15;
    const int half = lane >> 4;
    float* Srow = S + row * LP;
    const int c0 = half * (LP / 2);
    float mx = -INFINITY;
    for (int c = 0; c < LP / 2; ++c) mx = fmaxf(mx, Srow[c0 + c]);
    mx = fmaxf(mx, __shfl_xor(mx, 16, 32));
    if (!(mx > -INFINITY)) mx = 0.0f;        // fully-masked row guard
    float sum = 0.0f;
    for (int c = 0; c < LP / 2; ++c) {
      const float e = __expf(Srow[c0 + c] - mx);
      Srow[c0 + c] = e;
      sum += e;
    }
    sum += __shfl_xor(sum, 16, 32);
    const float inv = (sum > 0.0f) ? 1.0f / sum : 0.0f;
    _Float16* Prow = Ph + row * LP;
    for (int c = 0; c < LP / 2; ++c) Prow[c0 + c] = (_Float16)(Srow[c0 + c] * inv);
  }
  __syncthreads();

  // ---- pass 3: out = P @ V (K=32 steps over keys, 4 d-tiles) ---------------
  v8f o[4] = {};
  for (int t = 0; t < NT32; ++t) {
    const _Float16* prow = Ph + m * LP + t * 32;
    v16h pa;
    *(v8h*)&pa       = *(const v8h*)(prow + koffA);        // LDS b128 loads
    *((v8h*)&pa + 1) = *(const v8h*)(prow + 16 + koffA);
#pragma unroll
    for (int sub = 0; sub < 4; ++sub) {
      const _Float16* vrow = vt + ((size_t)bh * HD + sub * 16 + m) * LP + t * 32;
      v16h vb = load_b16(vrow, koffB);
      o[sub] = __builtin_amdgcn_wmma_f32_16x16x32_f16(
          false, pa, false, vb, (short)0, o[sub], false, false);
    }
  }

#pragma unroll
  for (int sub = 0; sub < 4; ++sub) {
#pragma unroll
    for (int r = 0; r < 8; ++r) {
      const int row = qt * 16 + r + mOff;
      const int d   = sub * 16 + m;
      y[((size_t)b * LP + row) * CDIM + h * HD + d] = (_Float16)o[sub][r];
    }
  }
#endif
}

// ---------------------------------------------------------------------------
// GEMM2: out = y @ W_proj (f32 out, drop padded rows)
// ---------------------------------------------------------------------------
__global__ __launch_bounds__(32) void k_gemm_out(const _Float16* __restrict__ yh,
                                                 const _Float16* __restrict__ wpt,
                                                 float* __restrict__ out) {
#ifdef CDNA5_DEVICE
  const int mt = blockIdx.x;          // 0..1343
  const int ng = blockIdx.y;          // 0..11
  const int lane = threadIdx.x;
  const int m = lane & 15;
  const int koffA = (lane < 16) ? 0 : 8;
  const int koffB = (lane < 16) ? 0 : 16;
  const int mOff  = (lane < 16) ? 0 : 8;

  const _Float16* arow = yh + (size_t)(mt * 16 + m) * CDIM;
  const _Float16* wcol = wpt + (size_t)(ng * 64 + m) * CDIM;

  v8f c[4] = {};
#pragma unroll 4
  for (int kk = 0; kk < CDIM; kk += 32) {
    v16h a = load_a16(arow + kk, koffA);
#pragma unroll
    for (int sub = 0; sub < 4; ++sub) {
      v16h b = load_b16(wcol + (size_t)sub * 16 * CDIM + kk, koffB);
      c[sub] = __builtin_amdgcn_wmma_f32_16x16x32_f16(
          false, a, false, b, (short)0, c[sub], false, false);
    }
  }

#pragma unroll
  for (int sub = 0; sub < 4; ++sub) {
#pragma unroll
    for (int r = 0; r < 8; ++r) {
      const int grow = mt * 16 + r + mOff;
      const int b  = grow / LP;
      const int rr = grow % LP;
      if (rr < LSEQ) {
        const int n = ng * 64 + sub * 16 + m;
        out[((size_t)b * LSEQ + rr) * CDIM + n] = c[sub][r];
      }
    }
  }
#endif
}

// ---------------------------------------------------------------------------
extern "C" void kernel_launch(void* const* d_in, const int* in_sizes, int n_in,
                              void* d_out, int out_size, void* d_ws, size_t ws_size,
                              hipStream_t stream) {
  const float* x  = (const float*)d_in[0];
  const float* Wa = (const float*)d_in[1];
  const float* Wp = (const float*)d_in[2];
  // d_in[3] = mem_size (int, ==128) -- baked in as MEMK
  float* out = (float*)d_out;
  char* ws = (char*)d_ws;

  _Float16* xh  = (_Float16*)(ws + OFF_XH);
  _Float16* wat = (_Float16*)(ws + OFF_WAT);
  _Float16* wpt = (_Float16*)(ws + OFF_WPT);
  _Float16* qh  = (_Float16*)(ws + OFF_QH);
  _Float16* kh  = (_Float16*)(ws + OFF_KH);
  _Float16* vt  = (_Float16*)(ws + OFF_VT);
  _Float16* yb  = (_Float16*)(ws + OFF_Y);

  {
    const size_t n = (size_t)BATCH * LP * CDIM;
    k_cvt_x<<<(unsigned)((n + 255) / 256), 256, 0, stream>>>(x, xh);
  }
  {
    const size_t n = (size_t)QKV * CDIM;
    k_cvt_wattn<<<(unsigned)((n + 255) / 256), 256, 0, stream>>>(Wa, wat);
  }
  {
    const size_t n = (size_t)CDIM * CDIM;
    k_cvt_wproj<<<(unsigned)((n + 255) / 256), 256, 0, stream>>>(Wp, wpt);
  }

  k_gemm_qkv<<<dim3(BATCH * LP / 16, QKV / 64), 32, 0, stream>>>(xh, wat, qh, kh, vt);
  k_attn<<<BATCH * NH * NT16, 32, 0, stream>>>(qh, kh, vt, yb);
  k_gemm_out<<<dim3(BATCH * LP / 16, CDIM / 64), 32, 0, stream>>>(yb, wpt, out);
}
